// EdgeLLMMambaLayer_69715909149349
// MI455X (gfx1250) — matrix operand order repaired and checked by reading, not compiled, not run on hardware
//
#include <hip/hip_runtime.h>
#include <hip/hip_bf16.h>
#include <cstdint>
#include <cstddef>

// ---------------- problem constants ----------------
#define S_    2048
#define HID_  4096
#define H_    128
#define P_    64
#define G_    8
#define I_    8192
#define N_    128
#define CD_   10240
#define KC_   4
#define GS_   (I_ / G_)            // 1024
#define PROJ_ (I_ + CD_ + H_)      // 18560
#define EPS_  1e-5f

// ---------------- CDNA5 WMMA / vector types ----------------
typedef __attribute__((ext_vector_type(16))) __bf16 v16bf;
typedef __attribute__((ext_vector_type(8)))  __bf16 bf16x8;
typedef __attribute__((ext_vector_type(4)))  __bf16 bf16x4;
typedef __attribute__((ext_vector_type(8)))  float  v8f;
typedef __attribute__((ext_vector_type(4)))  float  f32x4;

__device__ __forceinline__ bf16x8 cvt_bf8(f32x4 a, f32x4 b) {
    bf16x4 lo = __builtin_convertvector(a, bf16x4);   // native v_cvt f32->bf16
    bf16x4 hi = __builtin_convertvector(b, bf16x4);
    return __builtin_shufflevector(lo, hi, 0, 1, 2, 3, 4, 5, 6, 7);
}

__device__ __forceinline__ float silu_f(float v) {
    return v / (1.f + __expf(-v));
}

// ---------------------------------------------------------------------------
// WMMA GEMM:  C[M x N] = A[M x Kd] * W[N x Kd]^T   (row-major fp32 in, bf16
// WMMA fp32-accumulate). Block tile 64x128, BK=32, 256 threads / 8 waves.
// blockIdx.x = M-block (fastest) so consecutive blocks share the same weight
// slab through L2; the activation matrix stays L2-resident.
// ---------------------------------------------------------------------------
struct alignas(16) GemmTiles {
    __bf16 As[64][40];    // 40 bf16 = 80B row stride (16B multiple)
    __bf16 Bs[128][40];
};
union alignas(16) GemmSh {
    GemmTiles t;          // 15360 B
    float cbuf[64][128];  // 32768 B (epilogue staging)
};

__global__ __launch_bounds__(256)
void wmma_gemm_nt(const float* __restrict__ A,
                  const float* __restrict__ W,
                  float* __restrict__ C,
                  int M, int N, int Kd) {
    constexpr int BM = 64, BN = 128, BK = 32;
    __shared__ GemmSh sh;

    const int t    = threadIdx.x;
    const int lane = t & 31;
    const int wid  = t >> 5;
    const int waveM = (wid >> 2) * 32;   // 0,32
    const int waveN = (wid & 3) * 32;    // 0,32,64,96
    const int bm = blockIdx.x * BM;      // M fastest -> weight-slab reuse in L2
    const int bn = blockIdx.y * BN;

    v8f acc[2][2] = {};

    const int lm  = lane & 15;
    const int kgA = (lane >> 4) * 8;     // A frag k-group: 0 or 8
    const int kbB = (lane >> 4) * 8;     // B frag k-group: 0 or 8

    const int arow = t >> 2, aseg = (t & 3) * 8;    // A: 64 rows x 32 k, 8 f32/thread
    const int brow = t >> 1, bseg = (t & 1) * 16;   // B: 128 rows x 32 k, 16 f32/thread

    const float* aptr = &A[(size_t)(bm + arow) * Kd + aseg];
    const float* bptr = &W[(size_t)(bn + brow) * Kd + bseg];

    // ---- software pipeline: prefetch k-tile 0 into registers ----
    f32x4 ra0 = ((const f32x4*)aptr)[0];
    f32x4 ra1 = ((const f32x4*)aptr)[1];
    f32x4 rb0 = ((const f32x4*)bptr)[0];
    f32x4 rb1 = ((const f32x4*)bptr)[1];
    f32x4 rb2 = ((const f32x4*)bptr)[2];
    f32x4 rb3 = ((const f32x4*)bptr)[3];

    for (int k0 = 0; k0 < Kd; k0 += BK) {
        // ---- stage current tile (fp32 -> bf16, native cvt) into LDS ----
        {
            bf16x8* asd = (bf16x8*)__builtin_assume_aligned(&sh.t.As[arow][aseg], 16);
            asd[0] = cvt_bf8(ra0, ra1);
            bf16x8* bsd = (bf16x8*)__builtin_assume_aligned(&sh.t.Bs[brow][bseg], 16);
            bsd[0] = cvt_bf8(rb0, rb1);
            bsd[1] = cvt_bf8(rb2, rb3);
        }
        __syncthreads();

        // ---- issue next tile's global loads (overlap with WMMA below) ----
        if (k0 + BK < Kd) {
            const f32x4* an = (const f32x4*)(aptr + k0 + BK);
            const f32x4* bb = (const f32x4*)(bptr + k0 + BK);
            ra0 = an[0]; ra1 = an[1];
            rb0 = bb[0]; rb1 = bb[1]; rb2 = bb[2]; rb3 = bb[3];
        }

        // ---- assemble fragments per documented CDNA5 bf16 layouts ----
        bf16x8 a[2][2], b[2][2];
#pragma unroll
        for (int i = 0; i < 2; ++i) {
            const int m = waveM + i * 16 + lm;
            a[i][0] = *(const bf16x8*)__builtin_assume_aligned(&sh.t.As[m][kgA], 16);      // K=kg..kg+7
            a[i][1] = *(const bf16x8*)__builtin_assume_aligned(&sh.t.As[m][kgA + 16], 16); // K=kg+16..kg+23
        }
#pragma unroll
        for (int j = 0; j < 2; ++j) {
            const int n = waveN + j * 16 + lm;
            b[j][0] = *(const bf16x8*)__builtin_assume_aligned(&sh.t.Bs[n][kbB], 16);      // K=kb..kb+7
            b[j][1] = *(const bf16x8*)__builtin_assume_aligned(&sh.t.Bs[n][kbB + 16], 16); // K=kb+16..kb+23
        }

#pragma unroll
        for (int i = 0; i < 2; ++i)
#pragma unroll
            for (int j = 0; j < 2; ++j) {
                v16bf av = __builtin_shufflevector(a[i][0], a[i][1],
                    0, 1, 2, 3, 4, 5, 6, 7, 8, 9, 10, 11, 12, 13, 14, 15);
                v16bf bv = __builtin_shufflevector(b[j][0], b[j][1],
                    0, 1, 2, 3, 4, 5, 6, 7, 8, 9, 10, 11, 12, 13, 14, 15);
                acc[i][j] = __builtin_amdgcn_wmma_f32_16x16x32_bf16(
                    false, av, false, bv, (short)0, acc[i][j], false, false);
            }
        __syncthreads();
    }

    // ---- epilogue: stage 64x128 fp32 block in LDS, coalesced b128 stores ----
    const int rbase = (lane >> 4) * 8;   // VGPR r holds M = r + 8*(lane>=16), N = lane%16
#pragma unroll
    for (int i = 0; i < 2; ++i)
#pragma unroll
        for (int j = 0; j < 2; ++j)
#pragma unroll
            for (int r = 0; r < 8; ++r)
                sh.cbuf[waveM + i * 16 + rbase + r][waveN + j * 16 + lm] = acc[i][j][r];
    __syncthreads();

    float* cptr = C + (size_t)bm * N + bn;
#pragma unroll
    for (int it = 0; it < 8; ++it) {
        const int linear = it * 1024 + t * 4;     // 64x128 floats, 4 per thread
        const int row = linear >> 7;
        const int col = linear & 127;
        *(f32x4*)&cptr[(size_t)row * N + col] =
            *(const f32x4*)__builtin_assume_aligned(&sh.cbuf[row][col], 16);
    }
}

// ---------------------------------------------------------------------------
// Depthwise causal conv (K=4) over hbc slice of proj + SiLU; also emits
// conv_state_out (last 4 timesteps of raw hbc).
// ---------------------------------------------------------------------------
__global__ __launch_bounds__(256)
void conv_silu_k(const float* __restrict__ proj,
                 const float* __restrict__ conv_state,
                 const float* __restrict__ conv_w,
                 const float* __restrict__ conv_b,
                 float* __restrict__ xbc,
                 float* __restrict__ cso) {
    const int c = blockIdx.x * 256 + threadIdx.x;   // 0..CD-1
    const int s = blockIdx.y;                       // 0..S-1
    float acc = conv_b[c];
#pragma unroll
    for (int k = 0; k < KC_; ++k) {
        const int tt = s + k;   // position in x_full (prefix length K-1 = 3)
        float x = (tt < 3) ? conv_state[c * KC_ + tt + 1]
                           : proj[(size_t)(tt - 3) * PROJ_ + I_ + c];
        acc += x * conv_w[c * KC_ + k];
    }
    xbc[(size_t)s * CD_ + c] = silu_f(acc);

    if (s == 0) {
#pragma unroll
        for (int k = 0; k < KC_; ++k)
            cso[c * KC_ + k] = proj[(size_t)(S_ - KC_ + k) * PROJ_ + I_ + c];
    }
}

// ---------------------------------------------------------------------------
// dt = softplus(dt_raw + dt_bias)
// ---------------------------------------------------------------------------
__global__ __launch_bounds__(256)
void dt_softplus_k(const float* __restrict__ proj,
                   const float* __restrict__ dt_bias,
                   float* __restrict__ dtw) {
    const int idx = blockIdx.x * 256 + threadIdx.x;  // < S*H
    const int s = idx >> 7;
    const int h = idx & (H_ - 1);
    float v = proj[(size_t)s * PROJ_ + I_ + CD_ + h] + dt_bias[h];
    dtw[idx] = (v > 20.f) ? v : log1pf(__expf(v));
}

// ---------------------------------------------------------------------------
// Selective-state scan: one workgroup per head. State (64x128 f32) held in
// VGPRs: 256 threads x 32 floats. Per step, x/B/C broadcast through LDS.
// ---------------------------------------------------------------------------
__global__ __launch_bounds__(256)
void ssm_scan(const float* __restrict__ xbc,
              const float* __restrict__ dtw,
              const float* __restrict__ ssm_in,
              const float* __restrict__ A_log,
              const float* __restrict__ Dv,
              float* __restrict__ yb,
              float* __restrict__ ssm_out) {
    const int h  = blockIdx.x;        // head 0..127
    const int t  = threadIdx.x;       // 0..255
    const int p  = t & (P_ - 1);      // 0..63
    const int n0 = (t >> 6) * 32;     // 0,32,64,96
    const int g  = h / (H_ / G_);     // group = h/16

    __shared__ float sx[P_];
    __shared__ float sB[N_];
    __shared__ float sC[N_];
    __shared__ float red[256];

    float st[32];
    const float* s0 = ssm_in + ((size_t)h * P_ + p) * N_ + n0;
#pragma unroll
    for (int j = 0; j < 32; ++j) st[j] = s0[j];

    const float Ah = -__expf(A_log[h]);
    const float Dh = Dv[h];

    for (int s = 0; s < S_; ++s) {
        const size_t base = (size_t)s * CD_;
        if (t < P_)  sx[t]       = xbc[base + h * P_ + t];
        if (t < N_)  sB[t]       = xbc[base + I_ + g * N_ + t];
        else         sC[t - N_]  = xbc[base + I_ + G_ * N_ + g * N_ + (t - N_)];
        __syncthreads();

        const float dt = dtw[(size_t)s * H_ + h];
        const float dA = __expf(dt * Ah);
        const float xs = dt * sx[p];
        float partial = 0.f;
#pragma unroll
        for (int j = 0; j < 32; ++j) {
            st[j] = st[j] * dA + xs * sB[n0 + j];
            partial += st[j] * sC[n0 + j];
        }
        red[t] = partial;
        __syncthreads();
        if (t < P_) {
            float y = red[t] + red[t + 64] + red[t + 128] + red[t + 192] + Dh * sx[t];
            yb[(size_t)s * I_ + h * P_ + t] = y;
        }
        __syncthreads();
    }

    float* so = ssm_out + ((size_t)h * P_ + p) * N_ + n0;
#pragma unroll
    for (int j = 0; j < 32; ++j) so[j] = st[j];
}

// ---------------------------------------------------------------------------
// gated = y * silu(gate); grouped RMS norm over 1024-element groups.
// ---------------------------------------------------------------------------
__global__ __launch_bounds__(256)
void gated_norm(const float* __restrict__ yb,
                const float* __restrict__ proj,
                const float* __restrict__ nw,
                float* __restrict__ normed) {
    const int s    = blockIdx.x;
    const int gidx = blockIdx.y;
    const int t    = threadIdx.x;
    const int i0   = gidx * GS_ + t * 4;

    float4 yv = *(const float4*)&yb[(size_t)s * I_ + i0];
    float4 gv = *(const float4*)&proj[(size_t)s * PROJ_ + i0];
    float4 gd;
    gd.x = yv.x * silu_f(gv.x);
    gd.y = yv.y * silu_f(gv.y);
    gd.z = yv.z * silu_f(gv.z);
    gd.w = yv.w * silu_f(gv.w);

    __shared__ float red[256];
    red[t] = gd.x * gd.x + gd.y * gd.y + gd.z * gd.z + gd.w * gd.w;
    __syncthreads();
    for (int o = 128; o > 0; o >>= 1) {
        if (t < o) red[t] += red[t + o];
        __syncthreads();
    }
    const float scale = rsqrtf(red[0] / (float)GS_ + EPS_);

    float4 ov;
    ov.x = gd.x * scale * nw[i0 + 0];
    ov.y = gd.y * scale * nw[i0 + 1];
    ov.z = gd.z * scale * nw[i0 + 2];
    ov.w = gd.w * scale * nw[i0 + 3];
    *(float4*)&normed[(size_t)s * I_ + i0] = ov;
}

// ---------------------------------------------------------------------------
extern "C" void kernel_launch(void* const* d_in, const int* in_sizes, int n_in,
                              void* d_out, int out_size, void* d_ws, size_t ws_size,
                              hipStream_t stream) {
    const float* hidden     = (const float*)d_in[0];
    const float* conv_state = (const float*)d_in[1];
    const float* ssm_state  = (const float*)d_in[2];
    const float* W_in       = (const float*)d_in[3];
    const float* W_out      = (const float*)d_in[4];
    const float* conv_w     = (const float*)d_in[5];
    const float* conv_b     = (const float*)d_in[6];
    const float* A_log      = (const float*)d_in[7];
    const float* Dval       = (const float*)d_in[8];
    const float* dt_bias    = (const float*)d_in[9];
    const float* norm_w     = (const float*)d_in[10];

    float* out = (float*)d_out;                          // [S, HID]
    float* cso = out + (size_t)S_ * HID_;                // [CD, K]
    float* sso = cso + (size_t)CD_ * KC_;                // [H, P, N]

    float* ws     = (float*)d_ws;
    float* proj   = ws;                                  // [S, PROJ]
    float* xbc    = proj + (size_t)S_ * PROJ_;           // [S, CD]
    float* dtw    = xbc  + (size_t)S_ * CD_;             // [S, H]
    float* yb     = dtw  + (size_t)S_ * H_;              // [S, I]
    float* normed = yb   + (size_t)S_ * I_;              // [S, I]

    // 1) projected = hidden @ W_in.T   (M=2048, N=18560, K=4096); M-blocks fastest
    wmma_gemm_nt<<<dim3(S_ / 64, PROJ_ / 128), 256, 0, stream>>>(
        hidden, W_in, proj, S_, PROJ_, HID_);

    // 2) depthwise conv + SiLU, conv_state_out
    conv_silu_k<<<dim3(CD_ / 256, S_), 256, 0, stream>>>(
        proj, conv_state, conv_w, conv_b, xbc, cso);

    // 3) dt = softplus(dt + bias)
    dt_softplus_k<<<(S_ * H_) / 256, 256, 0, stream>>>(proj, dt_bias, dtw);

    // 4) sequential SSM scan (one WG per head)
    ssm_scan<<<H_, 256, 0, stream>>>(xbc, dtw, ssm_state, A_log, Dval, yb, sso);

    // 5) gated RMS norm
    gated_norm<<<dim3(S_, G_), 256, 0, stream>>>(yb, proj, norm_w, normed);

    // 6) output = normed @ W_out.T    (M=2048, N=4096, K=8192); M-blocks fastest
    wmma_gemm_nt<<<dim3(S_ / 64, HID_ / 128), 256, 0, stream>>>(
        normed, W_out, out, S_, HID_, I_);
}
